// SelfAttention_71442486001984
// MI455X (gfx1250) — matrix operand order repaired and checked by reading, not compiled, not run on hardware
//
#include <hip/hip_runtime.h>

typedef __attribute__((ext_vector_type(16))) __bf16 v16bf;
typedef __attribute__((ext_vector_type(8)))  __bf16 v8bf;
typedef __attribute__((ext_vector_type(8)))  float  v8f;

static constexpr int B_  = 2;
static constexpr int L_  = 2048;
static constexpr int DIM_ = 1024;
static constexpr int H_  = 16;
static constexpr int D_  = 64;
static constexpr int MROWS = B_ * L_;     // 4096
static constexpr int NQKV  = 3 * DIM_;    // 3072

static __device__ __forceinline__ v16bf cat8(v8bf lo, v8bf hi) {
  return __builtin_shufflevector(lo, hi, 0,1,2,3,4,5,6,7,8,9,10,11,12,13,14,15);
}

static __device__ __forceinline__ v8f wmma_bf16(v16bf a, v16bf b, v8f c) {
  // D = A(16x32 bf16) * B(32x16 bf16) + C(16x16 f32)
  return __builtin_amdgcn_wmma_f32_16x16x32_bf16(false, a, false, b, (short)0, c,
                                                 false, false);
}

// Async DMA: copy 16 bytes global -> LDS, tracked by ASYNCcnt.
// LDS hardware address = low 32 bits of the flat pointer (ISA 10.2).
static __device__ __forceinline__ void async_copy16(const __bf16* g, void* l) {
  unsigned lds = (unsigned)(unsigned long long)l;
  asm volatile("global_load_async_to_lds_b128 %0, %1, off"
               :: "v"(lds), "v"(g) : "memory");
}

static __device__ __forceinline__ void wait_async(int pending) {
  if (pending) asm volatile("s_wait_asynccnt 3" ::: "memory");
  else         asm volatile("s_wait_asynccnt 0" ::: "memory");
}

// ---------------- elementwise cast / transpose ----------------

__global__ void cast_f32_bf16_kernel(const float* __restrict__ in,
                                     __bf16* __restrict__ out, int n) {
  int i = blockIdx.x * blockDim.x + threadIdx.x;
  if (i < n) out[i] = (__bf16)in[i];
}

// in: [K, N] row-major f32  ->  out: [N, K] row-major bf16
__global__ void transpose_cast_kernel(const float* __restrict__ in,
                                      __bf16* __restrict__ out, int K, int N) {
  int i = blockIdx.x * blockDim.x + threadIdx.x;
  if (i < K * N) {
    int k = i / N, n = i % N;
    out[(size_t)n * K + k] = (__bf16)in[i];
  }
}

// ---------------- bf16 WMMA GEMM, async-LDS double-buffered ----------------
// C[M,N] (f32) = A[M,K](bf16,row-major) * Bt[N,K](bf16,row-major)^T + bias
// 256 threads = 8 waves (4M x 2N); block tile 64x128; wave tile 16x64; BK=32.
__global__ void __launch_bounds__(256) gemm_bf16_kernel(
    const __bf16* __restrict__ A, const __bf16* __restrict__ Bt,
    const float* __restrict__ bias, float* __restrict__ C,
    int M, int N, int K) {
  constexpr int BM = 64, BN = 128, BK = 32;
  __shared__ __align__(64) __bf16 sA[2][BM][BK];   // 8 KB
  __shared__ __align__(64) __bf16 sB[2][BN][BK];   // 16 KB
  const int tid  = threadIdx.x;
  const int lane = tid & 31;
  const int wave = tid >> 5;
  const int half = lane >> 4;
  const int l15  = lane & 15;
  const int wm   = wave & 3;        // 0..3 -> M sub-tile
  const int wn   = wave >> 2;       // 0..1 -> N group of 64
  const int mbase = blockIdx.y * BM;
  const int nbase = blockIdx.x * BN;
  const int akb = half ? 16 : 0;    // A frag: lane-half K half
  const int bkb = half ? 16 : 0;    // B frag: lane-half K half

  // Stage one 64x32 A tile (1 chunk/thread) + 128x32 B tile (2 chunks/thread).
  auto issue = [&](int buf, int k0) {
    {
      const int row = tid >> 2, ch = (tid & 3) * 8;
      async_copy16(A + (size_t)(mbase + row) * K + k0 + ch, &sA[buf][row][ch]);
    }
#pragma unroll
    for (int i = 0; i < 2; ++i) {
      const int c = tid + i * 256;
      const int row = c >> 2, ch = (c & 3) * 8;
      async_copy16(Bt + (size_t)(nbase + row) * K + k0 + ch, &sB[buf][row][ch]);
    }
  };

  v8f acc[4] = {};
  issue(0, 0);
  const int nk = K / BK;
  for (int i = 0; i < nk; ++i) {
    const int buf = i & 1;
    const int has_next = (i + 1 < nk);
    if (has_next) issue(buf ^ 1, (i + 1) * BK);
    wait_async(has_next);            // this wave's tile-i copies complete
    __syncthreads();                 // ... and everyone else's

    v8bf alo = *(const v8bf*)&sA[buf][wm * 16 + l15][akb];
    v8bf ahi = *(const v8bf*)&sA[buf][wm * 16 + l15][akb + 8];
    v16bf a  = cat8(alo, ahi);
#pragma unroll
    for (int t = 0; t < 4; ++t) {    // one A fragment feeds 4 WMMAs
      v16bf b = *(const v16bf*)&sB[buf][wn * 64 + t * 16 + l15][bkb];
      acc[t] = wmma_bf16(a, b, acc[t]);
    }
    __syncthreads();                 // safe to overwrite buf next iteration
  }

  const int mrow = mbase + wm * 16 + half * 8;
#pragma unroll
  for (int t = 0; t < 4; ++t) {
    const int ncol = nbase + wn * 64 + t * 16 + l15;
    const float bv = bias ? bias[ncol] : 0.f;
#pragma unroll
    for (int r = 0; r < 8; ++r)
      C[(size_t)(mrow + r) * N + ncol] = acc[t][r] + bv;
  }
}

// ---------------- RMSNorm + RoPE + layout change ----------------
// 1 wave per (b,l,h); lane t owns the RoPE pair (2t, 2t+1).
// qkv: [B*L, 3*DIM] f32; pe: [B,1,L,D/2,2,2] f32
// q,k out: [B,H,L,D] bf16 ; v out transposed: [B,H,D,L] bf16
__global__ void __launch_bounds__(32) norm_rope_kernel(
    const float* __restrict__ qkv, const float* __restrict__ pe,
    const float* __restrict__ qs, const float* __restrict__ ks,
    __bf16* __restrict__ qo, __bf16* __restrict__ ko, __bf16* __restrict__ vT) {
  const int bid = blockIdx.x;          // b*L*H + l*H + h
  const int h = bid % H_;
  const int l = (bid / H_) % L_;
  const int b = bid / (H_ * L_);
  const int t = threadIdx.x;           // 0..31
  const int d0 = 2 * t, d1 = d0 + 1;
  const size_t base = (size_t)(b * L_ + l) * NQKV;
  float q0 = qkv[base + (0 * H_ + h) * D_ + d0];
  float q1 = qkv[base + (0 * H_ + h) * D_ + d1];
  float k0 = qkv[base + (1 * H_ + h) * D_ + d0];
  float k1 = qkv[base + (1 * H_ + h) * D_ + d1];
  float v0 = qkv[base + (2 * H_ + h) * D_ + d0];
  float v1 = qkv[base + (2 * H_ + h) * D_ + d1];
  float sq = q0 * q0 + q1 * q1;
  float sk = k0 * k0 + k1 * k1;
#pragma unroll
  for (int m = 1; m < 32; m <<= 1) {
    sq += __shfl_xor(sq, m, 32);
    sk += __shfl_xor(sk, m, 32);
  }
  const float rq = rsqrtf(sq * (1.f / D_) + 1e-6f);
  const float rk = rsqrtf(sk * (1.f / D_) + 1e-6f);
  q0 *= rq * qs[d0]; q1 *= rq * qs[d1];
  k0 *= rk * ks[d0]; k1 *= rk * ks[d1];
  const size_t peb = ((size_t)(b * L_ + l) * (D_ / 2) + t) * 4;
  const float p00 = pe[peb + 0], p01 = pe[peb + 1];
  const float p10 = pe[peb + 2], p11 = pe[peb + 3];
  const float qe = p00 * q0 + p01 * q1, qoo = p10 * q0 + p11 * q1;
  const float ke = p00 * k0 + p01 * k1, koo = p10 * k0 + p11 * k1;
  const size_t qkb = ((size_t)(b * H_ + h) * L_ + l) * D_;
  qo[qkb + d0] = (__bf16)qe;  qo[qkb + d1] = (__bf16)qoo;
  ko[qkb + d0] = (__bf16)ke;  ko[qkb + d1] = (__bf16)koo;
  const size_t vb = (size_t)(b * H_ + h) * D_;
  vT[(vb + d0) * L_ + l] = (__bf16)v0;
  vT[(vb + d1) * L_ + l] = (__bf16)v1;
}

// ---------------- flash attention ----------------
// grid = B*H*(L/64); block = 128 (4 waves); wave owns a 16-query tile.
// Q,K: [B,H,L,D] bf16 ; Vt: [B,H,D,L] bf16 ; O: [B,L,DIM] bf16
__global__ void __launch_bounds__(128) flash_attn_kernel(
    const __bf16* __restrict__ Q, const __bf16* __restrict__ K,
    const __bf16* __restrict__ Vt, __bf16* __restrict__ O) {
  const int nqb  = L_ / 64;
  const int qblk = blockIdx.x % nqb;
  const int bh   = blockIdx.x / nqb;
  const int b    = bh / H_, h = bh % H_;
  const int wave = threadIdx.x >> 5;
  const int lane = threadIdx.x & 31;
  const int half = lane >> 4;
  const int l15  = lane & 15;
  const int akb  = half ? 16 : 0;
  const int bkb  = half ? 16 : 0;
  const int q0row = qblk * 64 + wave * 16;

  const __bf16* Qb = Q  + (size_t)bh * L_ * D_;
  const __bf16* Kb = K  + (size_t)bh * L_ * D_;
  const __bf16* Vb = Vt + (size_t)bh * D_ * L_;

  // Q A-fragments: 16x64 = two 16x32 chunks
  const __bf16* qrow = Qb + (size_t)(q0row + l15) * D_;
  v16bf aq[2];
#pragma unroll
  for (int c = 0; c < 2; ++c) {
    v8bf lo = *(const v8bf*)(qrow + c * 32 + akb);
    v8bf hi = *(const v8bf*)(qrow + c * 32 + akb + 8);
    aq[c] = cat8(lo, hi);
  }

  v8f acc[4] = {};                 // O accumulator: 16x64 as 4 n-tiles
  float rowm[8], rowl[8];
#pragma unroll
  for (int r = 0; r < 8; ++r) { rowm[r] = -1e30f; rowl[r] = 0.f; }

  __shared__ __align__(64) __bf16 ldsP[4][16][32];   // per-wave P staging
  __bf16 (*myP)[32] = ldsP[wave];

  for (int j = 0; j < L_; j += 32) {
    v8f s0 = {}, s1 = {};
#pragma unroll
    for (int c = 0; c < 2; ++c) {      // K-dim (=D) chunks of 32
      v16bf bk0 = *(const v16bf*)(Kb + (size_t)(j + l15) * D_ + c * 32 + bkb);
      v16bf bk1 = *(const v16bf*)(Kb + (size_t)(j + 16 + l15) * D_ + c * 32 + bkb);
      s0 = wmma_bf16(aq[c], bk0, s0);
      s1 = wmma_bf16(aq[c], bk1, s1);
    }
    // online softmax over the 32 new keys; row stats live per lane-half
#pragma unroll
    for (int r = 0; r < 8; ++r) {
      float x0 = s0[r] * 0.125f;       // 1/sqrt(64)
      float x1 = s1[r] * 0.125f;
      float mx = fmaxf(x0, x1);
#pragma unroll
      for (int msk = 1; msk < 16; msk <<= 1)
        mx = fmaxf(mx, __shfl_xor(mx, msk, 32));
      float mnew  = fmaxf(rowm[r], mx);
      float alpha = __expf(rowm[r] - mnew);
      float e0 = __expf(x0 - mnew);
      float e1 = __expf(x1 - mnew);
      float rs = e0 + e1;
#pragma unroll
      for (int msk = 1; msk < 16; msk <<= 1)
        rs += __shfl_xor(rs, msk, 32);
      rowl[r] = rowl[r] * alpha + rs;
      rowm[r] = mnew;
      myP[half * 8 + r][l15]      = (__bf16)e0;   // C-layout -> LDS
      myP[half * 8 + r][16 + l15] = (__bf16)e1;
      acc[0][r] *= alpha; acc[1][r] *= alpha;
      acc[2][r] *= alpha; acc[3][r] *= alpha;
    }
    // read P back in A-fragment layout (DS ops in-order within a wave)
    v8bf plo = *(const v8bf*)&myP[l15][akb];
    v8bf phi = *(const v8bf*)&myP[l15][akb + 8];
    v16bf ap = cat8(plo, phi);
#pragma unroll
    for (int t = 0; t < 4; ++t) {      // O += P(16x32) * V(32x16) per d-tile
      v16bf bv = *(const v16bf*)(Vb + (size_t)(t * 16 + l15) * L_ + j + bkb);
      acc[t] = wmma_bf16(ap, bv, acc[t]);
    }
  }

  float inv[8];
#pragma unroll
  for (int r = 0; r < 8; ++r) inv[r] = 1.f / rowl[r];
#pragma unroll
  for (int t = 0; t < 4; ++t) {
#pragma unroll
    for (int r = 0; r < 8; ++r) {
      const size_t q = (size_t)q0row + half * 8 + r;
      O[((size_t)b * L_ + q) * DIM_ + h * D_ + t * 16 + l15] =
          (__bf16)(acc[t][r] * inv[r]);
    }
  }
}

// ---------------- launcher ----------------

extern "C" void kernel_launch(void* const* d_in, const int* in_sizes, int n_in,
                              void* d_out, int out_size, void* d_ws, size_t ws_size,
                              hipStream_t stream) {
  const float* x       = (const float*)d_in[0];
  const float* pe      = (const float*)d_in[1];
  const float* w_qkv   = (const float*)d_in[2];
  const float* q_scale = (const float*)d_in[3];
  const float* k_scale = (const float*)d_in[4];
  const float* w_proj  = (const float*)d_in[5];
  const float* b_proj  = (const float*)d_in[6];
  float* out = (float*)d_out;

  char* ws = (char*)d_ws;
  size_t off = 0;
  auto alloc = [&](size_t bytes) -> char* {
    char* p = ws + off;
    off += (bytes + 255) & ~(size_t)255;
    return p;
  };
  __bf16* xb     = (__bf16*)alloc((size_t)MROWS * DIM_ * 2);
  __bf16* wqkvT  = (__bf16*)alloc((size_t)NQKV * DIM_ * 2);
  __bf16* wprojT = (__bf16*)alloc((size_t)DIM_ * DIM_ * 2);
  float*  qkv    = (float*) alloc((size_t)MROWS * NQKV * 4);
  __bf16* qb     = (__bf16*)alloc((size_t)MROWS * DIM_ * 2);
  __bf16* kb     = (__bf16*)alloc((size_t)MROWS * DIM_ * 2);
  __bf16* vT     = (__bf16*)alloc((size_t)MROWS * DIM_ * 2);
  __bf16* ob     = (__bf16*)alloc((size_t)MROWS * DIM_ * 2);
  (void)ws_size; (void)in_sizes; (void)n_in; (void)out_size;

  // 1. casts / transposes
  {
    int n = MROWS * DIM_;
    cast_f32_bf16_kernel<<<(n + 255) / 256, 256, 0, stream>>>(x, xb, n);
  }
  {
    int n = DIM_ * NQKV;
    transpose_cast_kernel<<<(n + 255) / 256, 256, 0, stream>>>(w_qkv, wqkvT, DIM_, NQKV);
  }
  {
    int n = DIM_ * DIM_;
    transpose_cast_kernel<<<(n + 255) / 256, 256, 0, stream>>>(w_proj, wprojT, DIM_, DIM_);
  }
  // 2. fused QKV projection: [4096,1024] x [1024,3072]
  {
    dim3 grid(NQKV / 128, MROWS / 64);
    gemm_bf16_kernel<<<grid, 256, 0, stream>>>(xb, wqkvT, nullptr, qkv,
                                               MROWS, NQKV, DIM_);
  }
  // 3. RMSNorm + RoPE + layout shuffle
  norm_rope_kernel<<<B_ * L_ * H_, 32, 0, stream>>>(qkv, pe, q_scale, k_scale,
                                                    qb, kb, vT);
  // 4. attention
  flash_attn_kernel<<<B_ * H_ * (L_ / 64), 128, 0, stream>>>(qb, kb, vT, ob);
  // 5. output projection + bias: [4096,1024] x [1024,1024]
  {
    dim3 grid(DIM_ / 128, MROWS / 64);
    gemm_bf16_kernel<<<grid, 256, 0, stream>>>(ob, wprojT, b_proj, out,
                                               MROWS, DIM_, DIM_);
  }
}
